// DanceDecoder_45363444580653
// MI455X (gfx1250) — matrix operand order repaired and checked by reading, not compiled, not run on hardware
//
#include <hip/hip_runtime.h>
#include <hip/hip_bf16.h>
#include <stddef.h>

// ---------------------------------------------------------------------------
// DanceDecoder (2-layer LSTM, B=8192, T=60, HID=512, OUT=51, LAT=64)
// Persistent per-workgroup recurrence: 256 WGs x 32 batch rows, all h/c state
// in LDS for all 60 steps. Weights converted once to bf16 and pre-swizzled
// into the CDNA5 WMMA B-operand register layout (one coalesced 1KB wave read
// per tile, straight from L2 into VGPRs). Math = v_wmma_f32_16x16x32_bf16.
// N-tiles processed in pairs to halve LDS A-operand traffic.
// ---------------------------------------------------------------------------

#define HID   512
#define NOUT  51
#define LAT   64
#define TSTEP 60
#define MROWS 32          // batch rows per workgroup
#define NWAVE 8

#define HSTR  520         // padded LDS row stride (halves) for h buffers
#define XSTR  72          // padded LDS row stride (halves) for x/z buffers

typedef __bf16 bf16_t;
typedef __attribute__((ext_vector_type(16))) __bf16 v16bf;
typedef __attribute__((ext_vector_type(8)))  __bf16 v8bf;
typedef __attribute__((ext_vector_type(8)))  float  v8f;

union V16U { v16bf v; v8bf h[2]; };

// LDS byte offsets (total 278016 B < 320KB WGP LDS).
// Double buffers are contiguous: ptr(sel) = base + sel*size.
#define OFF_H1   0           // 2 x 33280
#define H1_SZ    33280
#define OFF_H2   66560       // 2 x 33280
#define H2_SZ    33280
#define OFF_C1   133120      // 65536
#define OFF_C2   198656      // 65536
#define OFF_X    264192      // 2 x 4608
#define X_SZ     4608
#define OFF_Z    273408      // 4608
#define SMEM_BYTES 278016

// workspace byte offsets (bf16 weights, pre-swizzled; fused biases)
#define WS_FIW   0           // 1024x64  bf16  (131072 B)
#define WS_WIH0  131072      // 2048x64  bf16  (262144 B)
#define WS_WHH0  393216      // 2048x512 bf16  (2 MB)
#define WS_WIH1  2490368
#define WS_WHH1  4587520
#define WS_FOW   6684672     // 64x512   bf16  (65536 B)
#define WS_B0    6750208     // 2048 f32
#define WS_B1    6758400     // 2048 f32
// total ~6.77 MB of d_ws

__device__ __forceinline__ v8f wmma_bf16(v16bf a, v16bf b, v8f c) {
  return __builtin_amdgcn_wmma_f32_16x16x32_bf16(
      /*neg_a=*/false, a, /*neg_b=*/false, b,
      /*c_mod=*/(short)0, c, /*reuse_a=*/false, /*reuse_b=*/false);
}

// A-operand: 16x32 bf16 tile, lane L holds row M=L&15.
// elements 0..7  = K[kbase + base8 .. +8)
// elements 8..15 = K[kbase + 16 + base8 .. +8), base8 = (L>=16)?8:0
__device__ __forceinline__ v16bf load_a16(const bf16_t* rowk, int base8) {
  V16U u;
  u.h[0] = *(const v8bf*)(rowk + base8);
  u.h[1] = *(const v8bf*)(rowk + 16 + base8);
  return u.v;
}

__device__ __forceinline__ float sigmoidf_(float x) {
  return 1.0f / (1.0f + __expf(-x));
}

// ---------------------------------------------------------------------------
// Prep: fp32 -> bf16 conversion + swizzle into WMMA B-operand tile layout.
// Tiled layout: element index = ((nt*(K/32)+kc)*32 + lane)*16 + j
// maps to logical (n = nt*16 + (lane&15), k = kc*32 + (lane>=16?8:0) + (j&7) + (j>=8?16:0)).
// Pads n/k beyond source dims with zeros.
// ---------------------------------------------------------------------------
__device__ __forceinline__ void conv_tiled(const float* __restrict__ src,
                                           bf16_t* __restrict__ dst,
                                           int N, int K, int srcN, int srcK,
                                           size_t tid, size_t stride) {
  const size_t total = (size_t)N * K;
  const int kcn = K >> 5;
  for (size_t i = tid; i < total; i += stride) {
    const int j = (int)(i & 15);
    const int L = (int)((i >> 4) & 31);
    const size_t tile = i >> 9;
    const int kc = (int)(tile % kcn);
    const int nt = (int)(tile / kcn);
    const int n = nt * 16 + (L & 15);
    const int k = kc * 32 + ((L >> 4) ? 8 : 0) + (j & 7) + ((j >> 3) ? 16 : 0);
    float v = (n < srcN && k < srcK) ? src[(size_t)n * srcK + k] : 0.0f;
    dst[i] = (bf16_t)v;
  }
}

__global__ void dance_prep_kernel(
    const float* __restrict__ fiw,  const float* __restrict__ wih0,
    const float* __restrict__ whh0, const float* __restrict__ bih0,
    const float* __restrict__ bhh0, const float* __restrict__ wih1,
    const float* __restrict__ whh1, const float* __restrict__ bih1,
    const float* __restrict__ bhh1, const float* __restrict__ fow,
    bf16_t* o_fiw, bf16_t* o_wih0, bf16_t* o_whh0, bf16_t* o_wih1,
    bf16_t* o_whh1, bf16_t* o_fow, float* o_b0, float* o_b1) {
  const size_t tid = (size_t)blockIdx.x * blockDim.x + threadIdx.x;
  const size_t stride = (size_t)gridDim.x * blockDim.x;
  conv_tiled(fiw,  o_fiw,  2 * HID, LAT, 2 * HID, LAT,  tid, stride);
  conv_tiled(wih0, o_wih0, 4 * HID, 64,  4 * HID, NOUT, tid, stride);  // pad K 51->64
  conv_tiled(whh0, o_whh0, 4 * HID, HID, 4 * HID, HID,  tid, stride);
  conv_tiled(wih1, o_wih1, 4 * HID, HID, 4 * HID, HID,  tid, stride);
  conv_tiled(whh1, o_whh1, 4 * HID, HID, 4 * HID, HID,  tid, stride);
  conv_tiled(fow,  o_fow,  64, HID, NOUT, HID,          tid, stride);  // pad N 51->64
  for (size_t i = tid; i < 4 * HID; i += stride) {
    o_b0[i] = bih0[i] + bhh0[i];
    o_b1[i] = bih1[i] + bhh1[i];
  }
}

// ---------------------------------------------------------------------------
// One LSTM layer for this wave's slice: 16 rows (mt) x 128 hidden (nq),
// all four gates, n-tiles in pairs. gates = x@Wih^T + h@Whh^T + bias,
// then pointwise update of c (f32, LDS) and h (bf16, LDS).
// ---------------------------------------------------------------------------
__device__ __forceinline__ void lstm_layer(
    const bf16_t* xin, int xstr, int kcx,                    // x input (LDS)
    const bf16_t* hin,                                       // h input (LDS, HSTR)
    const bf16_t* __restrict__ wih, const bf16_t* __restrict__ whh,
    const float* __restrict__ bias,
    float* cbuf, bf16_t* hout,
    int mt, int nq, int lr, int hi, int lane) {
  const int mrow  = mt * 16 + lr;
  const int base8 = hi * 8;
  const size_t lane16 = (size_t)lane * 16;
  for (int hp = 0; hp < 4; ++hp) {            // pairs of n-tiles
    const int nt0 = nq * 8 + hp * 2;          // first tile of pair (per gate)
    const int nh0 = nq * 128 + hp * 32 + lr;  // hidden col of tile0 (N=lr)
    v8f acc[2][4];
    #pragma unroll
    for (int u = 0; u < 2; ++u) {
      #pragma unroll
      for (int g = 0; g < 4; ++g) {
        const float b = bias[g * HID + nh0 + u * 16];
        acc[u][g] = (v8f){b, b, b, b, b, b, b, b};
      }
    }
    // x contribution (K = kcx*32)
    #pragma unroll
    for (int kc = 0; kc < kcx; ++kc) {
      v16bf a = load_a16(xin + (size_t)mrow * xstr + kc * 32, base8);
      #pragma unroll
      for (int u = 0; u < 2; ++u) {
        #pragma unroll
        for (int g = 0; g < 4; ++g) {
          v16bf b = *(const v16bf*)(
              wih + (((size_t)(g * 32 + nt0 + u) * kcx + kc) << 9) + lane16);
          acc[u][g] = wmma_bf16(a, b, acc[u][g]);
        }
      }
    }
    // h contribution (K = 512 -> 16 chunks)
    #pragma unroll 2
    for (int kc = 0; kc < 16; ++kc) {
      v16bf a = load_a16(hin + (size_t)mrow * HSTR + kc * 32, base8);
      #pragma unroll
      for (int u = 0; u < 2; ++u) {
        #pragma unroll
        for (int g = 0; g < 4; ++g) {
          v16bf b = *(const v16bf*)(
              whh + (((size_t)(g * 32 + nt0 + u) * 16 + kc) << 9) + lane16);
          acc[u][g] = wmma_bf16(a, b, acc[u][g]);
        }
      }
    }
    // pointwise LSTM update; C layout: element r -> row M = r + 8*hi, col N=lr
    #pragma unroll
    for (int u = 0; u < 2; ++u) {
      const int nh = nh0 + u * 16;
      #pragma unroll
      for (int r = 0; r < 8; ++r) {
        const int m = mt * 16 + base8 + r;
        const float iv = sigmoidf_(acc[u][0][r]);
        const float fv = sigmoidf_(acc[u][1][r]);
        const float gv = tanhf(acc[u][2][r]);
        const float ov = sigmoidf_(acc[u][3][r]);
        const size_t ci = (size_t)m * HID + nh;
        const float cn = fv * cbuf[ci] + iv * gv;
        cbuf[ci] = cn;
        hout[(size_t)m * HSTR + nh] = (bf16_t)(ov * tanhf(cn));
      }
    }
  }
}

__global__ __launch_bounds__(256, 1) void dance_decoder_kernel(
    const float* __restrict__ z, const float* __restrict__ start_token,
    const bf16_t* __restrict__ fiw, const float* __restrict__ fib,
    const bf16_t* __restrict__ wih0, const bf16_t* __restrict__ whh0,
    const float* __restrict__ bias0, const bf16_t* __restrict__ wih1,
    const bf16_t* __restrict__ whh1, const float* __restrict__ bias1,
    const bf16_t* __restrict__ fow, const float* __restrict__ fob,
    float* __restrict__ out) {
  extern __shared__ char smem[];
  // Double-buffer pointers computed arithmetically from the parity bit —
  // no pointer arrays of LDS-derived addresses (lld can't relocate those).
  float* c1 = (float*)(smem + OFF_C1);
  float* c2 = (float*)(smem + OFF_C2);
  bf16_t* zb = (bf16_t*)(smem + OFF_Z);

  const int lane = (int)(threadIdx.x & 31);
  const int wave = (int)(threadIdx.x >> 5);
  const int lr = lane & 15;
  const int hi = lane >> 4;
  const int base8 = hi * 8;
  const size_t lane16 = (size_t)lane * 16;
  const int mt = wave >> 2;    // 0..1 : batch sub-tile
  const int nq = wave & 3;     // 0..3 : hidden-unit quarter
  const size_t wg_m0 = (size_t)blockIdx.x * MROWS;

  // stage z (bf16) and x0 = start_token into LDS
  {
    bf16_t* x0 = (bf16_t*)(smem + OFF_X);
    for (int idx = threadIdx.x; idx < MROWS * 64; idx += 256) {
      const int m = idx >> 6, j = idx & 63;
      zb[(size_t)m * XSTR + j] = (bf16_t)z[(wg_m0 + m) * LAT + j];
      x0[(size_t)m * XSTR + j] = (bf16_t)((j < NOUT) ? start_token[j] : 0.0f);
    }
  }
  __syncthreads();

  // init: h_init = z @ fc_init_w^T + fc_init_b  -> h0 (cols 0..511), c0 (512..1023)
  for (int task = wave; task < 128; task += NWAVE) {
    const int imt = task & 1;
    const int nt = task >> 1;          // 0..63 (uniform per wave iteration)
    const int ncol = nt * 16 + lr;
    const float bv = fib[ncol];
    v8f acc = {bv, bv, bv, bv, bv, bv, bv, bv};
    const int mrow = imt * 16 + lr;
    #pragma unroll
    for (int kc = 0; kc < 2; ++kc) {
      v16bf a = load_a16(zb + (size_t)mrow * XSTR + kc * 32, base8);
      v16bf b = *(const v16bf*)(fiw + (((size_t)nt * 2 + kc) << 9) + lane16);
      acc = wmma_bf16(a, b, acc);
    }
    bf16_t* h1_0 = (bf16_t*)(smem + OFF_H1);
    bf16_t* h2_0 = (bf16_t*)(smem + OFF_H2);
    // nt < 32 is wave-uniform (boundary at a multiple of 16): scalar branch,
    // no per-lane exec masking.
    if (nt < 32) {
      #pragma unroll
      for (int r = 0; r < 8; ++r) {
        const int m = imt * 16 + base8 + r;
        const bf16_t hv = (bf16_t)acc[r];
        h1_0[(size_t)m * HSTR + ncol] = hv;
        h2_0[(size_t)m * HSTR + ncol] = hv;
      }
    } else {
      #pragma unroll
      for (int r = 0; r < 8; ++r) {
        const int m = imt * 16 + base8 + r;
        c1[(size_t)m * HID + (ncol - HID)] = acc[r];
        c2[(size_t)m * HID + (ncol - HID)] = acc[r];
      }
    }
  }
  __syncthreads();

  int p = 0;
  for (int t = 0; t < TSTEP; ++t) {
    const int q = p ^ 1;
    bf16_t* h1_cur = (bf16_t*)(smem + OFF_H1 + (size_t)p * H1_SZ);
    bf16_t* h1_new = (bf16_t*)(smem + OFF_H1 + (size_t)q * H1_SZ);
    bf16_t* h2_cur = (bf16_t*)(smem + OFF_H2 + (size_t)p * H2_SZ);
    bf16_t* h2_new = (bf16_t*)(smem + OFF_H2 + (size_t)q * H2_SZ);
    bf16_t* x_cur  = (bf16_t*)(smem + OFF_X  + (size_t)p * X_SZ);
    bf16_t* x_new  = (bf16_t*)(smem + OFF_X  + (size_t)q * X_SZ);

    // layer 0: x_cur, h1_cur -> h1_new, c1
    lstm_layer(x_cur, XSTR, 2, h1_cur, wih0, whh0, bias0, c1, h1_new,
               mt, nq, lr, hi, lane);
    __syncthreads();
    // layer 1: h1_new (as x), h2_cur -> h2_new, c2
    lstm_layer(h1_new, HSTR, 16, h2_cur, wih1, whh1, bias1, c2, h2_new,
               mt, nq, lr, hi, lane);
    __syncthreads();
    // fc_out: frame = h2_new @ fc_out_w^T + b ; feeds x_new and d_out
    {
      const int nt = nq;                  // 4 n-tiles cover padded 64 cols
      const int ncol = nt * 16 + lr;
      const int mrow = mt * 16 + lr;
      v8f acc = {0.f, 0.f, 0.f, 0.f, 0.f, 0.f, 0.f, 0.f};
      #pragma unroll 4
      for (int kc = 0; kc < 16; ++kc) {
        v16bf a = load_a16(h2_new + (size_t)mrow * HSTR + kc * 32, base8);
        v16bf b = *(const v16bf*)(fow + (((size_t)nt * 16 + kc) << 9) + lane16);
        acc = wmma_bf16(a, b, acc);
      }
      const float bb = (ncol < NOUT) ? fob[ncol] : 0.0f;
      #pragma unroll
      for (int r = 0; r < 8; ++r) {
        const int m = mt * 16 + base8 + r;
        const float v = acc[r] + bb;
        x_new[(size_t)m * XSTR + ncol] = (bf16_t)v;   // padded cols -> exact 0
        if (ncol < NOUT)
          out[((wg_m0 + m) * TSTEP + t) * NOUT + ncol] = v;
      }
    }
    __syncthreads();
    p ^= 1;
  }
}

extern "C" void kernel_launch(void* const* d_in, const int* in_sizes, int n_in,
                              void* d_out, int out_size, void* d_ws, size_t ws_size,
                              hipStream_t stream) {
  (void)in_sizes; (void)n_in; (void)out_size; (void)ws_size;
  const float* z    = (const float*)d_in[0];
  const float* st   = (const float*)d_in[1];
  const float* fiw  = (const float*)d_in[2];
  const float* fib  = (const float*)d_in[3];
  const float* wih0 = (const float*)d_in[4];
  const float* whh0 = (const float*)d_in[5];
  const float* bih0 = (const float*)d_in[6];
  const float* bhh0 = (const float*)d_in[7];
  const float* wih1 = (const float*)d_in[8];
  const float* whh1 = (const float*)d_in[9];
  const float* bih1 = (const float*)d_in[10];
  const float* bhh1 = (const float*)d_in[11];
  const float* fow  = (const float*)d_in[12];
  const float* fob  = (const float*)d_in[13];

  char* ws = (char*)d_ws;               // needs ~6.8 MB scratch
  bf16_t* o_fiw  = (bf16_t*)(ws + WS_FIW);
  bf16_t* o_wih0 = (bf16_t*)(ws + WS_WIH0);
  bf16_t* o_whh0 = (bf16_t*)(ws + WS_WHH0);
  bf16_t* o_wih1 = (bf16_t*)(ws + WS_WIH1);
  bf16_t* o_whh1 = (bf16_t*)(ws + WS_WHH1);
  bf16_t* o_fow  = (bf16_t*)(ws + WS_FOW);
  float*  o_b0   = (float*)(ws + WS_B0);
  float*  o_b1   = (float*)(ws + WS_B1);

  dance_prep_kernel<<<2048, 256, 0, stream>>>(
      fiw, wih0, whh0, bih0, bhh0, wih1, whh1, bih1, bhh1, fow,
      o_fiw, o_wih0, o_whh0, o_wih1, o_whh1, o_fow, o_b0, o_b1);

  dance_decoder_kernel<<<8192 / MROWS, 256, SMEM_BYTES, stream>>>(
      z, st, o_fiw, fib, o_wih0, o_whh0, o_b0, o_wih1, o_whh1, o_b1,
      o_fow, fob, (float*)d_out);
}
